// AdaptiveModelV3_33157147525663
// MI455X (gfx1250) — matrix operand order repaired and checked by reading, not compiled, not run on hardware
//
#include <hip/hip_runtime.h>

typedef __bf16 bf16_t;
typedef __attribute__((ext_vector_type(16))) __bf16 v16bf;
typedef __attribute__((ext_vector_type(8)))  __bf16 v8bf;
typedef __attribute__((ext_vector_type(8)))  float  v8f;

#define B_    4096
#define S_    201
#define DIN_  64
#define H_    128
#define G4_   512   /* 4*H */
#define KC_   192   /* combined K = DIN + H */
#define KD_   32
#define DK_   32
#define DV_   32
#define DOUT_ 32
#define TEMP_ 0.1f
#define NOV_  0.5f
#define EPS_  1e-8f

/* workspace byte offsets */
#define OFF_WCB 0        /* 512*192 bf16 = 196608 */
#define OFF_KN  196608   /* 200*32 f32   = 25600  */
#define OFF_MB  222208   /* 200 f32      = 800    */
#define OFF_CTX 223232   /* 4096*32 f32  = 524288 */

/* branch-free transcendentals: hardware v_tanh_f32 when available */
#if __has_builtin(__builtin_amdgcn_tanhf)
__device__ __forceinline__ float tanh_fast(float x) { return __builtin_amdgcn_tanhf(x); }
#else
__device__ __forceinline__ float tanh_fast(float x) {
  return 1.0f - 2.0f / (1.0f + __expf(2.0f * x));
}
#endif
__device__ __forceinline__ float sigf(float x) { return 0.5f * tanh_fast(0.5f * x) + 0.5f; }

/* A-matrix (bf16 16x32, MxK) lane layout per ISA 7.12.2:
   lane L: M = L&15; K-half = L>>4.
   elements 0..7  -> K = (L>>4)*8 + e
   elements 8..15 -> K = 16 + (L>>4)*8 + (e-8)                      */
__device__ __forceinline__ v16bf a_from_f32(const float* p0, const float* p1) {
  v16bf a;
#pragma unroll
  for (int e = 0; e < 8; ++e) { a[e] = (bf16_t)p0[e]; a[8 + e] = (bf16_t)p1[e]; }
  return a;
}
__device__ __forceinline__ v16bf a_from_lds(const bf16_t* p0, const bf16_t* p1) {
  v8bf lo = *(const v8bf*)p0;
  v8bf hi = *(const v8bf*)p1;
  v16bf a;
#pragma unroll
  for (int e = 0; e < 8; ++e) { a[e] = lo[e]; a[8 + e] = hi[e]; }
  return a;
}

/* ---- prep: combined bf16 weights [W_ih | W_hh] (512x192), normalized slot
        keys with 1/TEMP folded, activity mask bias ---- */
__global__ void prep_kernel(const float* __restrict__ inputs,
                            const float* __restrict__ W_ih,
                            const float* __restrict__ W_hh,
                            const float* __restrict__ W_k,
                            const float* __restrict__ b_k,
                            bf16_t* __restrict__ Wcb,
                            float* __restrict__ kn,
                            float* __restrict__ mbias) {
  const int t = threadIdx.x;
  for (int i = t; i < G4_ * KC_; i += 256) {
    int n = i / KC_, k = i - n * KC_;
    float v = (k < DIN_) ? W_ih[n * DIN_ + k] : W_hh[n * H_ + (k - DIN_)];
    Wcb[i] = (bf16_t)v;
  }
  for (int i = t; i < 200 * DK_; i += 256) {
    int p = i >> 5, d = i & 31;
    float acc = b_k[d];
    const float* kp = inputs + p * DIN_;
#pragma unroll
    for (int k = 0; k < KD_; ++k) acc += kp[k] * W_k[d * KD_ + k];
    kn[i] = acc;
  }
  __syncthreads();
  for (int p = t; p < 200; p += 256) {
    float n2 = 0.f;
#pragma unroll
    for (int d = 0; d < DK_; ++d) { float v = kn[p * DK_ + d]; n2 += v * v; }
    float sc = 1.0f / ((sqrtf(n2) + EPS_) * TEMP_);
#pragma unroll
    for (int d = 0; d < DK_; ++d) kn[p * DK_ + d] *= sc;
    float vs = 0.f;
    const float* vp = inputs + p * DIN_ + KD_;
#pragma unroll
    for (int d = 0; d < DV_; ++d) vs += vp[d];
    mbias[p] = (vs >= NOV_) ? 0.0f : -1.0e9f;
  }
}

/* ---------------- attention: one wave per batch row ---------------- */
__global__ void __launch_bounds__(256)
attn_kernel(const float* __restrict__ inputs,
            const float* __restrict__ W_k,
            const float* __restrict__ b_k,
            const float* __restrict__ kn,
            const float* __restrict__ mbias,
            float* __restrict__ ctx) {
  __shared__ float qbuf[8][DK_];
  __shared__ float abuf[8][200];
  const int lane = threadIdx.x & 31;
  const int w    = threadIdx.x >> 5;
  const int b    = blockIdx.x * 8 + w;
  const float* xlast = inputs + ((size_t)b * S_ + (S_ - 1)) * DIN_;

  float q = b_k[lane];
#pragma unroll
  for (int k = 0; k < KD_; ++k) q += xlast[k] * W_k[lane * KD_ + k];
  float n2 = q * q;
#pragma unroll
  for (int off = 16; off > 0; off >>= 1) n2 += __shfl_xor(n2, off, 32);
  qbuf[w][lane] = q / (sqrtf(n2) + EPS_);
  __syncthreads();

  float lg[7];
  float mx = -3.4e38f;
#pragma unroll
  for (int j = 0; j < 7; ++j) {
    int p = lane + 32 * j;
    float l = -3.4e38f;
    if (p < 200) {
      l = mbias[p];
      const float* knp = kn + p * DK_;
#pragma unroll
      for (int d = 0; d < DK_; ++d) l += qbuf[w][d] * knp[d];
    }
    lg[j] = l;
    mx = fmaxf(mx, l);
  }
#pragma unroll
  for (int off = 16; off > 0; off >>= 1) mx = fmaxf(mx, __shfl_xor(mx, off, 32));
  float sum = 0.f;
#pragma unroll
  for (int j = 0; j < 7; ++j) {
    int p = lane + 32 * j;
    float e = (p < 200) ? __expf(lg[j] - mx) : 0.0f;
    lg[j] = e;
    sum += e;
  }
#pragma unroll
  for (int off = 16; off > 0; off >>= 1) sum += __shfl_xor(sum, off, 32);
  float inv = 1.0f / sum;
#pragma unroll
  for (int j = 0; j < 7; ++j) {
    int p = lane + 32 * j;
    if (p < 200) abuf[w][p] = lg[j] * inv;
  }
  __syncthreads();

  float c = 0.f;
  for (int p = 0; p < 200; ++p) c += abuf[w][p] * inputs[p * DIN_ + KD_ + lane];
  ctx[(size_t)b * DV_ + lane] = c;
}

/* ---------------- persistent LSTM + fused output ----------------
   16 batch rows per WG, 8 waves. Wave w owns h columns [w*16, w*16+16)
   of each gate. Weight B-tiles register-resident (24 x v16bf/wave);
   h state ping-pongs between two LDS buffers -> ONE barrier per step. */
__global__ void __launch_bounds__(256)
lstm_kernel(const float* __restrict__ inputs,
            const bf16_t* __restrict__ Wcb,
            const float* __restrict__ b_ih,
            const float* __restrict__ b_hh,
            const float* __restrict__ ctx,
            const float* __restrict__ W_o,
            const float* __restrict__ b_o,
            float* __restrict__ out) {
  __shared__ bf16_t hbuf[2][16][H_]; /* 8 KB: ping-pong h state (A source) */
  __shared__ float  hfin[16][H_];    /* 8 KB: final h, f32                 */

  const int tid  = threadIdx.x;
  const int lane = tid & 31;
  const int w    = tid >> 5;    /* wave 0..7 */
  const int b0   = blockIdx.x * 16;
  const int m    = lane & 15;   /* WMMA row within tile  */
  const int kh   = lane >> 4;   /* WMMA K-half selector  */

  for (int i = tid; i < 16 * H_; i += 256) ((bf16_t*)hbuf[0])[i] = (bf16_t)0.0f;

  /* register-resident B tiles + per-gate bias (as WMMA C vector);
     column n = gi*128 + w*16 + m */
  v16bf Bt[4][6];
  v8f   bwv[4];
#pragma unroll
  for (int gi = 0; gi < 4; ++gi) {
    const int n = gi * H_ + w * 16 + m;
    const float bv = b_ih[n] + b_hh[n];
#pragma unroll
    for (int r = 0; r < 8; ++r) bwv[gi][r] = bv;
    const bf16_t* wrow = Wcb + (size_t)n * KC_;
#pragma unroll
    for (int kb = 0; kb < 6; ++kb)
      Bt[gi][kb] = a_from_lds(wrow + kb * 32 + kh * 8, wrow + kb * 32 + 16 + kh * 8);
  }

  v8f cc; /* cell state in WMMA C layout */
#pragma unroll
  for (int r = 0; r < 8; ++r) cc[r] = 0.0f;

  const float* xrow = inputs + (size_t)(b0 + m) * S_ * DIN_;
  __syncthreads();

  for (int s = 0; s < S_; ++s) {
    const int pp = s & 1;
    const float* xs = xrow + (size_t)s * DIN_;
    v16bf a0 = a_from_f32(xs + kh * 8,      xs + 16 + kh * 8);
    v16bf a1 = a_from_f32(xs + 32 + kh * 8, xs + 48 + kh * 8);
    if (s + 1 < S_) __builtin_prefetch(xs + DIN_, 0, 1);

    v8f acc[4]; /* i, f, g, o */
#pragma unroll
    for (int gi = 0; gi < 4; ++gi)
      acc[gi] = __builtin_amdgcn_wmma_f32_16x16x32_bf16(
          false, a0, false, Bt[gi][0], (short)0, bwv[gi], false, false);
#pragma unroll
    for (int gi = 0; gi < 4; ++gi)
      acc[gi] = __builtin_amdgcn_wmma_f32_16x16x32_bf16(
          false, a1, false, Bt[gi][1], (short)0, acc[gi], false, false);

#pragma unroll
    for (int q = 0; q < 4; ++q) {
      v16bf ah = a_from_lds(&hbuf[pp][m][q * 32 + kh * 8],
                            &hbuf[pp][m][q * 32 + 16 + kh * 8]);
#pragma unroll
      for (int gi = 0; gi < 4; ++gi)
        acc[gi] = __builtin_amdgcn_wmma_f32_16x16x32_bf16(
            false, ah, false, Bt[gi][2 + q], (short)0, acc[gi], false, false);
    }

    const int hcol = w * 16 + m;
#pragma unroll
    for (int r = 0; r < 8; ++r) {
      float cv = sigf(acc[1][r]) * cc[r] + sigf(acc[0][r]) * tanh_fast(acc[2][r]);
      cc[r] = cv;
      float hv = sigf(acc[3][r]) * tanh_fast(cv);
      hbuf[pp ^ 1][r + 8 * kh][hcol] = (bf16_t)hv;
      if (s == S_ - 1) hfin[r + 8 * kh][hcol] = hv;
    }
    __syncthreads(); /* writes to hbuf[pp^1] visible before next step reads */
  }

  /* out = [h, ctx] @ W_o^T + b_o   (16x160 @ 160x32) */
  {
    const int d  = tid & 31;
    const int mb = tid >> 5;
#pragma unroll
    for (int k = 0; k < 2; ++k) {
      const int mm = mb + 8 * k;
      float acc = b_o[d];
      const float* wo = W_o + d * (H_ + DV_);
      for (int j = 0; j < H_; ++j) acc += hfin[mm][j] * wo[j];
      const float* cx = ctx + (size_t)(b0 + mm) * DV_;
#pragma unroll
      for (int j = 0; j < DV_; ++j) acc += cx[j] * wo[H_ + j];
      out[(size_t)(b0 + mm) * DOUT_ + d] = acc;
    }
  }
}

extern "C" void kernel_launch(void* const* d_in, const int* in_sizes, int n_in,
                              void* d_out, int out_size, void* d_ws, size_t ws_size,
                              hipStream_t stream) {
  const float* inputs = (const float*)d_in[0];
  const float* W_ih   = (const float*)d_in[1];
  const float* W_hh   = (const float*)d_in[2];
  const float* b_ih   = (const float*)d_in[3];
  const float* b_hh   = (const float*)d_in[4];
  const float* W_k    = (const float*)d_in[5];
  const float* b_k    = (const float*)d_in[6];
  const float* W_o    = (const float*)d_in[7];
  const float* b_o    = (const float*)d_in[8];

  char* ws = (char*)d_ws;
  bf16_t* Wcb = (bf16_t*)(ws + OFF_WCB);
  float*  kn  = (float*)(ws + OFF_KN);
  float*  mb  = (float*)(ws + OFF_MB);
  float*  ctx = (float*)(ws + OFF_CTX);
  float*  out = (float*)d_out;

  prep_kernel<<<1, 256, 0, stream>>>(inputs, W_ih, W_hh, W_k, b_k, Wcb, kn, mb);
  attn_kernel<<<B_ / 8, 256, 0, stream>>>(inputs, W_k, b_k, kn, mb, ctx);
  lstm_kernel<<<B_ / 16, 256, 0, stream>>>(inputs, Wcb, b_ih, b_hh, ctx,
                                           W_o, b_o, out);
}
